// MultiLevelGCN_46677704573012
// MI455X (gfx1250) — matrix operand order repaired and checked by reading, not compiled, not run on hardware
//
#include <hip/hip_runtime.h>
#include <hip/hip_bf16.h>

#define GN 16384
#define GF 256
#define N_IDX 8192
#define MLP_H 128
#define N_CLS 16

typedef _Float16 v16h __attribute__((ext_vector_type(16)));
typedef float    v8f  __attribute__((ext_vector_type(8)));

static constexpr float A_SCALE = 16384.0f;            // 2^14, exact
static constexpr float A_INV   = 1.0f / 16384.0f;     // 2^-14, exact

union Frag16 { uint4 u[2]; v16h h; };
union Pack8  { uint4 u; _Float16 h[8]; };

__device__ __forceinline__ v8f vzero8() {
    v8f z;
#pragma unroll
    for (int i = 0; i < 8; ++i) z[i] = 0.0f;
    return z;
}

// B operand (K x 16), 16-bit: lanes 0-15 hold row n0+lane with K=k0..k0+15,
// lanes 16-31 hold K=k0+16..k0+31. Storage is row-major [n][k] with stride ldb.
__device__ __forceinline__ v16h load_b_frag(const _Float16* __restrict__ B, int ldb,
                                            int n0, int k0, int lane) {
    const _Float16* p = B + (size_t)(n0 + (lane & 15)) * ldb + k0 + ((lane >> 4) << 4);
    Frag16 f;
    f.u[0] = *(const uint4*)(p);
    f.u[1] = *(const uint4*)(p + 8);
    return f.h;
}

// A operand 16x32, 16-bit, from row-major LDS tile (stride in halves, mult of 8):
// lane<16: row m0+lane, K 0..15; lane>=16: K 16..31.
__device__ __forceinline__ v16h load_a_frag_lds(const _Float16* S, int stride,
                                                int m0, int k0, int lane) {
    const _Float16* p = S + (m0 + (lane & 15)) * stride + k0 + ((lane >> 4) << 4);
    Frag16 f;
    f.u[0] = *(const uint4*)(p);
    f.u[1] = *(const uint4*)(p + 8);
    return f.h;
}

// Convert 16 f32 (scaled by 2^14) -> 16 f16 and store as two b128 LDS writes.
__device__ __forceinline__ void cvt_store16(_Float16* dst, float4 f0, float4 f1,
                                            float4 f2, float4 f3) {
    Pack8 a, b;
    a.h[0] = (_Float16)(f0.x * A_SCALE); a.h[1] = (_Float16)(f0.y * A_SCALE);
    a.h[2] = (_Float16)(f0.z * A_SCALE); a.h[3] = (_Float16)(f0.w * A_SCALE);
    a.h[4] = (_Float16)(f1.x * A_SCALE); a.h[5] = (_Float16)(f1.y * A_SCALE);
    a.h[6] = (_Float16)(f1.z * A_SCALE); a.h[7] = (_Float16)(f1.w * A_SCALE);
    b.h[0] = (_Float16)(f2.x * A_SCALE); b.h[1] = (_Float16)(f2.y * A_SCALE);
    b.h[2] = (_Float16)(f2.z * A_SCALE); b.h[3] = (_Float16)(f2.w * A_SCALE);
    b.h[4] = (_Float16)(f3.x * A_SCALE); b.h[5] = (_Float16)(f3.y * A_SCALE);
    b.h[6] = (_Float16)(f3.z * A_SCALE); b.h[7] = (_Float16)(f3.w * A_SCALE);
    *(uint4*)(dst)     = a.u;
    *(uint4*)(dst + 8) = b.u;
}

// One staged K=64 tile: 2 k-halves x (4 M-tiles x 2 N-tiles) = 16 WMMAs per wave.
__device__ __forceinline__ void gemm1_tile(const _Float16* Abuf,
                                           const _Float16* __restrict__ xT,
                                           v8f acc[8], int w, int lane, int k0) {
#pragma unroll
    for (int kk = 0; kk < 2; ++kk) {
        v16h af[4];
#pragma unroll
        for (int mt = 0; mt < 4; ++mt)
            af[mt] = load_a_frag_lds(Abuf, 72, mt * 16, kk * 32, lane);
#pragma unroll
        for (int nt = 0; nt < 2; ++nt) {
            v16h bf = load_b_frag(xT, GN, w * 32 + nt * 16, k0 + kk * 32, lane);
#pragma unroll
            for (int mt = 0; mt < 4; ++mt)
                acc[mt * 2 + nt] = __builtin_amdgcn_wmma_f32_16x16x32_f16(
                    false, af[mt], false, bf, (short)0, acc[mt * 2 + nt], false, false);
        }
    }
}

// ---------------- setup: f32 -> f16 conversions ----------------

__global__ __launch_bounds__(256) void convert_x0_kernel(const float* __restrict__ x0,
                                                         _Float16* __restrict__ xT) {
    int i = blockIdx.x * 256 + threadIdx.x;     // over N*F
    int m = i >> 8, f = i & 255;
    xT[(size_t)f * GN + m] = (_Float16)x0[i];
}

__global__ __launch_bounds__(256) void convert_w_kernel(
    const float* __restrict__ gw, const float* __restrict__ w1, const float* __restrict__ w2,
    _Float16* __restrict__ gwh, _Float16* __restrict__ w1h, _Float16* __restrict__ w2h) {
    int i = blockIdx.x * 256 + threadIdx.x;
    if (i < 3 * 256 * 256) gwh[i] = (_Float16)gw[i];
    if (i < MLP_H * 256)   w1h[i] = (_Float16)w1[i];
    if (i < N_CLS * MLP_H) w2h[i] = (_Float16)w2[i];
}

// ---------------- fused GCN layer: x' = relu((A @ x) @ W^T + b) ----------------
// Block: 256 threads (8 waves), computes 64 output rows x all 256 features.
// Wave w owns columns [w*32, w*32+32): 4 M-tiles x 2 N-tiles of 16x16 accumulators.
// A is streamed f32->f16 through a double-buffered LDS tile (K=64 per stage,
// one barrier per stage; next tile's global loads overlap current tile's WMMAs).

__global__ __launch_bounds__(256) void gcn_layer_kernel(
    const float* __restrict__ A, const _Float16* __restrict__ xT,
    const _Float16* __restrict__ Wl, const float* __restrict__ bias,
    _Float16* __restrict__ out, int transpose_out) {
    __shared__ _Float16 As[2][64 * 72];  // 2 x (64x64 f16 tile, padded stride 72)
    __shared__ _Float16 Ys[64 * 264];    // 64x256 y tile (f16), padded stride 264

    const int t = threadIdx.x;
    const int w = t >> 5, lane = t & 31;
    const int m0 = blockIdx.x * 64;
    const int ar = t >> 2;               // 0..63 : row in tile
    const int ac = (t & 3) << 4;         // 0,16,32,48 : col group (16 f32 each)

    v8f acc[8];
#pragma unroll
    for (int i = 0; i < 8; ++i) acc[i] = vzero8();

    const float* arow = A + (size_t)(m0 + ar) * GN + ac;
    _Float16* myAs[2] = { &As[0][ar * 72 + ac], &As[1][ar * 72 + ac] };

    // ---- GEMM1: y = (A * 2^14) @ x, K = 16384, software-pipelined ----
    {
        const float4* ap = (const float4*)(arow);
        cvt_store16(myAs[0], ap[0], ap[1], ap[2], ap[3]);
    }
    __syncthreads();

    int buf = 0;
    for (int k0 = 64; k0 < GN; k0 += 64) {
        __builtin_prefetch(arow + k0 + 64, 0, 0);
        const float4* ap = (const float4*)(arow + k0);
        float4 f0 = ap[0], f1 = ap[1], f2 = ap[2], f3 = ap[3];
        gemm1_tile(As[buf], xT, acc, w, lane, k0 - 64);
        cvt_store16(myAs[buf ^ 1], f0, f1, f2, f3);
        buf ^= 1;
        __syncthreads();
    }
    gemm1_tile(As[buf], xT, acc, w, lane, GN - 64);

    // ---- stage y (rescaled by 2^-14) into LDS as f16 ----
#pragma unroll
    for (int mt = 0; mt < 4; ++mt)
#pragma unroll
        for (int nt = 0; nt < 2; ++nt) {
            int n  = w * 32 + nt * 16 + (lane & 15);
            int mb = mt * 16 + ((lane >> 4) << 3);
#pragma unroll
            for (int r = 0; r < 8; ++r)
                Ys[(mb + r) * 264 + n] = (_Float16)(acc[mt * 2 + nt][r] * A_INV);
        }
    __syncthreads();

    // ---- GEMM2: z = y @ W^T, K = 256 ----
#pragma unroll
    for (int i = 0; i < 8; ++i) acc[i] = vzero8();
#pragma unroll
    for (int k0 = 0; k0 < 256; k0 += 32) {
        v16h af[4];
#pragma unroll
        for (int mt = 0; mt < 4; ++mt) af[mt] = load_a_frag_lds(Ys, 264, mt * 16, k0, lane);
#pragma unroll
        for (int nt = 0; nt < 2; ++nt) {
            v16h bf = load_b_frag(Wl, 256, w * 32 + nt * 16, k0, lane);
#pragma unroll
            for (int mt = 0; mt < 4; ++mt)
                acc[mt * 2 + nt] = __builtin_amdgcn_wmma_f32_16x16x32_f16(
                    false, af[mt], false, bf, (short)0, acc[mt * 2 + nt], false, false);
        }
    }

    // ---- bias + relu + store (transposed f16 for next layer, or row-major) ----
#pragma unroll
    for (int nt = 0; nt < 2; ++nt) {
        int n = w * 32 + nt * 16 + (lane & 15);
        float bv = bias[n];
#pragma unroll
        for (int mt = 0; mt < 4; ++mt) {
            int mb = m0 + mt * 16 + ((lane >> 4) << 3);
            if (transpose_out) {
                Pack8 st;
#pragma unroll
                for (int r = 0; r < 8; ++r)
                    st.h[r] = (_Float16)fmaxf(acc[mt * 2 + nt][r] + bv, 0.0f);
                *(uint4*)(out + (size_t)n * GN + mb) = st.u;
            } else {
#pragma unroll
                for (int r = 0; r < 8; ++r)
                    out[(size_t)(mb + r) * GF + n] =
                        (_Float16)fmaxf(acc[mt * 2 + nt][r] + bv, 0.0f);
            }
        }
    }
}

// ---------------- MLP head: gather -> relu(h W1^T + b1) -> softmax(e W2^T + b2) ----------------

__global__ __launch_bounds__(256) void mlp_kernel(
    const _Float16* __restrict__ x3, const int* __restrict__ idx,
    const _Float16* __restrict__ W1, const float* __restrict__ b1,
    const _Float16* __restrict__ W2, const float* __restrict__ b2,
    float* __restrict__ enc_out, float* __restrict__ prob_out) {
    __shared__ _Float16 Hs[64 * 264];   // gathered h tile 64x256
    __shared__ _Float16 Es[64 * 136];   // encode tile 64x128 (f16)

    const int t = threadIdx.x, w = t >> 5, lane = t & 31;
    const int m0 = blockIdx.x * 64;

    // gather 64 node rows into LDS
    {
        int r = t >> 2;
        int c = (t & 3) << 6;  // 0,64,128,192 halves
        const uint4* src = (const uint4*)(x3 + (size_t)idx[m0 + r] * GF + c);
        uint4* dst = (uint4*)(Hs + r * 264 + c);
#pragma unroll
        for (int i = 0; i < 8; ++i) dst[i] = src[i];
    }
    __syncthreads();

    // encode = relu(h @ W1^T + b1): wave w owns cols [w*16, w*16+16), K=256
    v8f acc[4];
#pragma unroll
    for (int i = 0; i < 4; ++i) acc[i] = vzero8();
#pragma unroll
    for (int k0 = 0; k0 < 256; k0 += 32) {
        v16h bf = load_b_frag(W1, 256, w * 16, k0, lane);
#pragma unroll
        for (int mt = 0; mt < 4; ++mt) {
            v16h af = load_a_frag_lds(Hs, 264, mt * 16, k0, lane);
            acc[mt] = __builtin_amdgcn_wmma_f32_16x16x32_f16(
                false, af, false, bf, (short)0, acc[mt], false, false);
        }
    }
    {
        int j = w * 16 + (lane & 15);
        float bv = b1[j];
#pragma unroll
        for (int mt = 0; mt < 4; ++mt) {
            int mb = mt * 16 + ((lane >> 4) << 3);
#pragma unroll
            for (int r = 0; r < 8; ++r) {
                float v = fmaxf(acc[mt][r] + bv, 0.0f);
                enc_out[(size_t)(m0 + mb + r) * MLP_H + j] = v;
                Es[(mb + r) * 136 + j] = (_Float16)v;
            }
        }
    }
    __syncthreads();

    // out = softmax(encode @ W2^T + b2): waves 0..3 each take one 16-row M-tile
    if (w < 4) {
        const int mt = w;
        v8f a2 = vzero8();
#pragma unroll
        for (int k0 = 0; k0 < MLP_H; k0 += 32) {
            v16h bf = load_b_frag(W2, MLP_H, 0, k0, lane);
            v16h af = load_a_frag_lds(Es, 136, mt * 16, k0, lane);
            a2 = __builtin_amdgcn_wmma_f32_16x16x32_f16(
                false, af, false, bf, (short)0, a2, false, false);
        }
        float bv2 = b2[lane & 15];
#pragma unroll
        for (int r = 0; r < 8; ++r) {
            float v = a2[r] + bv2;
            float mx = v;
#pragma unroll
            for (int msk = 1; msk < 16; msk <<= 1) mx = fmaxf(mx, __shfl_xor(mx, msk, 32));
            float e = __expf(v - mx);
            float s = e;
#pragma unroll
            for (int msk = 1; msk < 16; msk <<= 1) s += __shfl_xor(s, msk, 32);
            int m = m0 + mt * 16 + ((lane >> 4) << 3) + r;
            prob_out[(size_t)m * N_CLS + (lane & 15)] = e / s;
        }
    }
}

// ---------------- host launcher ----------------

extern "C" void kernel_launch(void* const* d_in, const int* in_sizes, int n_in,
                              void* d_out, int out_size, void* d_ws, size_t ws_size,
                              hipStream_t stream) {
    const float* A  = (const float*)d_in[0];
    const float* x0 = (const float*)d_in[1];
    const float* gw = (const float*)d_in[2];
    const float* gb = (const float*)d_in[3];
    const float* w1 = (const float*)d_in[4];
    const float* b1 = (const float*)d_in[5];
    const float* w2 = (const float*)d_in[6];
    const float* b2 = (const float*)d_in[7];
    const int*  idx = (const int*)d_in[8];

    char* ws = (char*)d_ws;
    _Float16* xTa = (_Float16*)(ws);                                  // 8 MB  [F][N]
    _Float16* xTb = (_Float16*)(ws + (size_t)8  * 1024 * 1024);       // 8 MB
    _Float16* x3  = (_Float16*)(ws + (size_t)16 * 1024 * 1024);       // 8 MB  row-major [N][F]
    _Float16* gwh = (_Float16*)(ws + (size_t)24 * 1024 * 1024);       // 384 KB
    _Float16* w1h = gwh + 3 * 256 * 256;                              // 64 KB
    _Float16* w2h = w1h + MLP_H * 256;                                // 4 KB

    float* enc  = (float*)d_out;                 // [8192][128]
    float* prob = enc + (size_t)N_IDX * MLP_H;   // [8192][16]

    convert_x0_kernel<<<(GN * GF) / 256, 256, 0, stream>>>(x0, xTa);
    convert_w_kernel<<<(3 * 256 * 256) / 256, 256, 0, stream>>>(gw, w1, w2, gwh, w1h, w2h);

    gcn_layer_kernel<<<GN / 64, 256, 0, stream>>>(A, xTa, gwh,              gb,       xTb, 1);
    gcn_layer_kernel<<<GN / 64, 256, 0, stream>>>(A, xTb, gwh + 256 * 256,  gb + 256, xTa, 1);
    gcn_layer_kernel<<<GN / 64, 256, 0, stream>>>(A, xTa, gwh + 2*256*256,  gb + 512, x3,  0);

    mlp_kernel<<<N_IDX / 64, 256, 0, stream>>>(x3, idx, w1h, b1, w2h, b2, enc, prob);
}
